// ODConvBlock_77369540870231
// MI455X (gfx1250) — matrix operand order repaired and checked by reading, not compiled, not run on hardware
//
#include <hip/hip_runtime.h>
#include <math.h>

#define B_  16
#define C_  192
#define O_  384
#define H_  128
#define W_  128
#define K_  4
#define A_  16
#define HW_ (H_*W_)
#define TS  64            // spatial tile along W per block
#define DWP 208           // padded row pitch (bf16 elems) for LDS dw tile

typedef __bf16 v16bf __attribute__((ext_vector_type(16)));
typedef float  v8f   __attribute__((ext_vector_type(8)));

union AFrag { v16bf v; int4 q[2]; };
union BFrag { v16bf v; int4 q[2]; };

static __device__ __forceinline__ unsigned short f2bf(float f) {
    unsigned int u = __builtin_bit_cast(unsigned int, f);
    u += 0x7FFFu + ((u >> 16) & 1u);          // round-to-nearest-even
    return (unsigned short)(u >> 16);
}
static __device__ __forceinline__ float sigmoidf_(float t) {
    return 1.0f / (1.0f + __expf(-t));
}

// ---------------- 1) Global average pool: pooled[B*C] ----------------
__global__ void odc_gap(const float* __restrict__ x, float* __restrict__ pooled) {
    int bc = blockIdx.x;                       // 0..3071
    const float* p = x + (size_t)bc * HW_;
    float s = 0.f;
    for (int i = threadIdx.x; i < HW_; i += 256) s += p[i];
    __shared__ float red[256];
    red[threadIdx.x] = s;
    __syncthreads();
    for (int o = 128; o > 0; o >>= 1) {
        if (threadIdx.x < o) red[threadIdx.x] += red[threadIdx.x + o];
        __syncthreads();
    }
    if (threadIdx.x == 0) pooled[bc] = red[0] * (1.0f / (float)HW_);
}

// ---------------- 2) Attention head: c_attn, s_attn, k_attn ----------------
__global__ void odc_attn(const float* __restrict__ pooled,
                         const float* __restrict__ attn_fc_w,
                         const float* __restrict__ bn_g, const float* __restrict__ bn_b,
                         const float* __restrict__ bn_m, const float* __restrict__ bn_v,
                         const float* __restrict__ ch_w, const float* __restrict__ ch_b,
                         const float* __restrict__ sp_w, const float* __restrict__ sp_b,
                         const float* __restrict__ kn_w, const float* __restrict__ kn_b,
                         float* __restrict__ c_attn, float* __restrict__ s_attn,
                         float* __restrict__ k_attn) {
    __shared__ float s_a[B_ * A_];
    int tid = threadIdx.x;
    { // a = relu(bn(pooled @ attn_fc_w.T)), one (b,j) per thread
        int b = tid >> 4, j = tid & 15;
        float t = 0.f;
        for (int c = 0; c < C_; ++c) t += pooled[b * C_ + c] * attn_fc_w[j * C_ + c];
        t = (t - bn_m[j]) * rsqrtf(bn_v[j] + 1e-5f) * bn_g[j] + bn_b[j];
        s_a[b * A_ + j] = fmaxf(t, 0.f);
    }
    __syncthreads();
    for (int idx = tid; idx < B_ * C_; idx += 256) {   // channel attention
        int b = idx / C_, c = idx % C_;
        float t = ch_b[c];
        for (int j = 0; j < A_; ++j) t += s_a[b * A_ + j] * ch_w[c * A_ + j];
        c_attn[idx] = sigmoidf_(t);
    }
    for (int idx = tid; idx < B_ * 9; idx += 256) {    // spatial attention
        int b = idx / 9, t9 = idx % 9;
        float t = sp_b[t9];
        for (int j = 0; j < A_; ++j) t += s_a[b * A_ + j] * sp_w[t9 * A_ + j];
        s_attn[idx] = sigmoidf_(t);
    }
    if (tid < B_) {                                    // kernel softmax (K=4)
        int b = tid;
        float v[K_], mx = -1e30f;
        for (int k = 0; k < K_; ++k) {
            float t = kn_b[k];
            for (int j = 0; j < A_; ++j) t += s_a[b * A_ + j] * kn_w[k * A_ + j];
            v[k] = t; mx = fmaxf(mx, t);
        }
        float sum = 0.f;
        for (int k = 0; k < K_; ++k) { v[k] = __expf(v[k] - mx); sum += v[k]; }
        float inv = 1.0f / sum;
        for (int k = 0; k < K_; ++k) k_attn[b * K_ + k] = v[k] * inv;
    }
}

// ---------------- 3) Fold attentions into per-(b,c) 3x3 weights ----------------
__global__ void odc_fold(const float* __restrict__ od_w,
                         const float* __restrict__ c_attn,
                         const float* __restrict__ s_attn,
                         const float* __restrict__ k_attn,
                         float* __restrict__ wfold) {
    int idx = blockIdx.x * 256 + threadIdx.x;          // 0..3071 : (b,c)
    if (idx >= B_ * C_) return;
    int b = idx / C_, c = idx % C_;
    float ca = c_attn[idx];
    for (int t = 0; t < 9; ++t) {
        float s = 0.f;
        for (int k = 0; k < K_; ++k) s += k_attn[b * K_ + k] * od_w[(k * C_ + c) * 9 + t];
        wfold[idx * 9 + t] = ca * s_attn[b * 9 + t] * s;
    }
}

// ---------------- 4) Fused BN scale/shift (per output channel) ----------------
__global__ void odc_scale(const float* __restrict__ pw_b,
                          const float* __restrict__ g, const float* __restrict__ be,
                          const float* __restrict__ m, const float* __restrict__ v,
                          float* __restrict__ scale, float* __restrict__ shift) {
    int o = threadIdx.x;
    if (o < O_) {
        float sc = g[o] * rsqrtf(v[o] + 1e-3f);
        scale[o] = sc;
        shift[o] = (pw_b[o] - m[o]) * sc + be[o];
    }
}

// ---------------- 5) Pre-pack pw_w into WMMA A-fragment register images ----------------
// Apack[((k_step*24 + m_sub)*32 + lane)*8 + dw]  (dwords, bf16 pairs, 6 k-steps x 24 m-subs)
// A 16x32 bf16 layout: lane<16: M=lane, dw0..3 = K {0,1..6,7}, dw4..7 = K {16..23};
//                      lane>=16 adds +8 to K base.  (ISA 7.12.2)
__global__ void odc_packA(const float* __restrict__ pw_w, unsigned int* __restrict__ Apack) {
    int kk   = blockIdx.x / 24;                 // k-step 0..5
    int msub = blockIdx.x % 24;                 // m-subtile 0..23
    int lane = threadIdx.x >> 3;                // 0..31
    int dw   = threadIdx.x & 7;                 // 0..7
    int L    = lane & 15, hi = lane >> 4;
    int kpair = ((dw < 4) ? (2 * dw) : (16 + 2 * (dw - 4))) + hi * 8;
    int c0 = kk * 32 + kpair;
    int o  = msub * 16 + L;
    unsigned int lo = f2bf(pw_w[o * C_ + c0]);
    unsigned int hi16 = f2bf(pw_w[o * C_ + c0 + 1]);
    Apack[((kk * 24 + msub) * 32 + lane) * 8 + dw] = lo | (hi16 << 16);
}

// ---------------- 6) Fused depthwise 3x3 + WMMA pointwise + BN ----------------
// grid: (W/TS, H, B), block 256 = 8 waves.
// GEMM per block: M=O=384 (24 subtiles), N=TS=64 (4 subtiles), K=C=192 (6 steps).
// Per wave: 6 m-subtiles x 2 n-subtiles -> 12 accumulators, 6 A-frags + 2 B-frags
// live per k-step so all 12 WMMAs run behind a single load-wait.
__global__ void __launch_bounds__(256)
odc_main(const float* __restrict__ x,
         const float* __restrict__ wfold,
         const unsigned int* __restrict__ Apack,
         const float* __restrict__ scale, const float* __restrict__ shift,
         float* __restrict__ out) {
    __shared__ float          s_wf[C_ * 9];
    __shared__ unsigned short s_dw[TS * DWP];   // dw tile bf16, [m][c] c-contiguous
    __shared__ float          s_sc[O_], s_sh[O_];

    const int tid = threadIdx.x;
    const int b  = blockIdx.z;
    const int h  = blockIdx.y;
    const int w0 = blockIdx.x * TS;

    for (int i = tid; i < C_ * 9; i += 256) s_wf[i] = wfold[b * C_ * 9 + i];
    for (int i = tid; i < O_; i += 256) { s_sc[i] = scale[i]; s_sh[i] = shift[i]; }
    __syncthreads();

    // ---- depthwise 3x3 (weights already carry channel/spatial/kernel attn) ----
    for (int idx = tid; idx < C_ * TS; idx += 256) {
        int c = idx >> 6, m = idx & (TS - 1);
        int w = w0 + m;
        const float* wf = &s_wf[c * 9];
        const float* xb = x + ((size_t)(b * C_ + c) * H_) * W_;
        float acc = 0.f;
        #pragma unroll
        for (int dy = -1; dy <= 1; ++dy) {
            int hh = h + dy;
            if ((unsigned)hh < (unsigned)H_) {
                const float* row = xb + (size_t)hh * W_;
                #pragma unroll
                for (int dx = -1; dx <= 1; ++dx) {
                    int ww = w + dx;
                    if ((unsigned)ww < (unsigned)W_)
                        acc += wf[(dy + 1) * 3 + (dx + 1)] * row[ww];
                }
            }
        }
        s_dw[m * DWP + c] = f2bf(acc);
    }
    __syncthreads();

    // ---- WMMA: D[o, w] += pw_w[o,c] * dw[c, w], bf16 in / f32 accumulate ----
    const int wave  = tid >> 5;
    const int lane  = tid & 31;
    const int l16   = lane & 15;
    const int hi    = lane >> 4;
    const int n_grp = wave & 1;                 // 2 spatial 16-col subtiles
    const int m_grp = wave >> 1;                // 6 o-subtiles: m_grp*6 ..

    v8f acc[6][2];
    #pragma unroll
    for (int mi = 0; mi < 6; ++mi)
        #pragma unroll
        for (int ni = 0; ni < 2; ++ni)
            #pragma unroll
            for (int i = 0; i < 8; ++i) acc[mi][ni][i] = 0.f;

    #pragma unroll
    for (int k = 0; k < 6; ++k) {
        // B fragments (32x16 bf16): lane<16 holds K=cb..cb+7, cb+16..cb+23 of its column; lane>=16: +8
        BFrag bfr[2];
        #pragma unroll
        for (int ni = 0; ni < 2; ++ni) {
            int p = (n_grp * 2 + ni) * 16 + l16;
            const int4* bsrc =
                reinterpret_cast<const int4*>(&s_dw[p * DWP + k * 32 + hi * 8]);
            bfr[ni].q[0] = bsrc[0];
            bfr[ni].q[1] = bsrc[2];
        }
        // A fragments: per-lane contiguous 32B register images, L2/WGP$-resident
        AFrag afr[6];
        #pragma unroll
        for (int mi = 0; mi < 6; ++mi) {
            const int4* asrc = reinterpret_cast<const int4*>(
                Apack + (((k * 24 + (m_grp * 6 + mi)) * 32 + lane) << 3));
            afr[mi].q[0] = asrc[0];
            afr[mi].q[1] = asrc[1];
        }
        // warm WGP$ with next k-step's A fragments while WMMAs execute
        if (k < 5) {
            const char* pf = (const char*)(Apack +
                ((((k + 1) * 24 + m_grp * 6) * 32 + lane) << 3));
            __builtin_prefetch(pf, 0, 1);
        }
        // 12 independent WMMAs behind one wait
        #pragma unroll
        for (int mi = 0; mi < 6; ++mi)
            #pragma unroll
            for (int ni = 0; ni < 2; ++ni)
                acc[mi][ni] = __builtin_amdgcn_wmma_f32_16x16x32_bf16(
                    false, afr[mi].v, false, bfr[ni].v, (short)0, acc[mi][ni],
                    false, false);
    }

    // ---- epilogue: fused BN, coalesced along w (N=lane%16) ----
    #pragma unroll
    for (int mi = 0; mi < 6; ++mi) {
        int obase = (m_grp * 6 + mi) * 16 + hi * 8;    // D: M = r + 8*hi
        #pragma unroll
        for (int ni = 0; ni < 2; ++ni) {
            int wcol = w0 + (n_grp * 2 + ni) * 16 + l16;
            #pragma unroll
            for (int r = 0; r < 8; ++r) {
                int o = obase + r;
                out[((size_t)(b * O_ + o) * H_ + h) * W_ + wcol] =
                    acc[mi][ni][r] * s_sc[o] + s_sh[o];
            }
        }
    }
}

// ---------------- host launcher ----------------
extern "C" void kernel_launch(void* const* d_in, const int* in_sizes, int n_in,
                              void* d_out, int out_size, void* d_ws, size_t ws_size,
                              hipStream_t stream) {
    const float* x        = (const float*)d_in[0];
    const float* od_w     = (const float*)d_in[1];
    const float* afc_w    = (const float*)d_in[2];
    const float* abn_g    = (const float*)d_in[3];
    const float* abn_b    = (const float*)d_in[4];
    const float* abn_m    = (const float*)d_in[5];
    const float* abn_v    = (const float*)d_in[6];
    const float* ch_w     = (const float*)d_in[7];
    const float* ch_b     = (const float*)d_in[8];
    const float* sp_w     = (const float*)d_in[9];
    const float* sp_b     = (const float*)d_in[10];
    const float* kn_w     = (const float*)d_in[11];
    const float* kn_b     = (const float*)d_in[12];
    const float* pw_w     = (const float*)d_in[13];
    const float* pw_b     = (const float*)d_in[14];
    const float* bn_g     = (const float*)d_in[15];
    const float* bn_b     = (const float*)d_in[16];
    const float* bn_m     = (const float*)d_in[17];
    const float* bn_v     = (const float*)d_in[18];
    float* out = (float*)d_out;

    // workspace carve-up (bytes)
    char* ws = (char*)d_ws;
    float* pooled  = (float*)(ws + 0);           // 12288
    float* c_attn  = (float*)(ws + 12288);       // 12288
    float* s_attn  = (float*)(ws + 24576);       // 576
    float* k_attn  = (float*)(ws + 25152);       // 256  (pad to 25600)
    float* wfold   = (float*)(ws + 25600);       // 110592
    float* scale   = (float*)(ws + 136192);      // 1536
    float* shift   = (float*)(ws + 137728);      // 1536
    unsigned int* Apack = (unsigned int*)(ws + 139264); // 147456 -> total 286720

    odc_gap<<<B_ * C_, 256, 0, stream>>>(x, pooled);
    odc_attn<<<1, 256, 0, stream>>>(pooled, afc_w, abn_g, abn_b, abn_m, abn_v,
                                    ch_w, ch_b, sp_w, sp_b, kn_w, kn_b,
                                    c_attn, s_attn, k_attn);
    odc_fold<<<(B_ * C_ + 255) / 256, 256, 0, stream>>>(od_w, c_attn, s_attn, k_attn, wfold);
    odc_scale<<<1, O_, 0, stream>>>(pw_b, bn_g, bn_b, bn_m, bn_v, scale, shift);
    odc_packA<<<6 * 24, 256, 0, stream>>>(pw_w, Apack);

    dim3 grid(W_ / TS, H_, B_);
    odc_main<<<grid, 256, 0, stream>>>(x, wfold, Apack, scale, shift, out);
}
